// TorchAttractorLanguageModel_41755672051866
// MI455X (gfx1250) — matrix operand order repaired and checked by reading, not compiled, not run on hardware
//
#include <hip/hip_runtime.h>
#include <hip/hip_bf16.h>
#include <math.h>

typedef __attribute__((ext_vector_type(16))) __bf16 v16bf;
typedef __attribute__((ext_vector_type(8)))  float  v8f;

static constexpr int   kD      = 512;
static constexpr int   kT      = 2048;
static constexpr int   kSteps  = 4;
static constexpr float kDt     = 0.04f;
static constexpr float kCubic  = 0.008f;
static constexpr float kAlpha  = 0.97f;
static constexpr float kWSlow  = 0.5f;
static constexpr float kEps    = 1e-6f;
static constexpr float kLnEps  = 1e-5f;

// Dynamic LDS layout for the scan kernel:
//   [0, 256KB)                : bf16 A-fragments for K-tiles 8..15 (pre-swizzled)
//   then 5 x 512 f32 vectors  : sS, sSlow, sSig, sY, sE
//   then 512 bf16             : sSbf (bf16 shadow of sS)
//   then 8 f32                : sRed
static constexpr int kABytes   = 32 * 8 * 32 * 16 * 2;  // mt*kt*lane*16bf16 = 262144
static constexpr int kSmemSize = kABytes + 5 * kD * 4 + kD * 2 + 8 * 4;

__device__ __forceinline__ float waveAllSum(float v) {
#pragma unroll
  for (int o = 16; o > 0; o >>= 1) v += __shfl_xor(v, o, 32);
  return v;
}

// ---------------------------------------------------------------------------
// Sequential attractor scan: one 1024-thread workgroup (32 wave32 waves).
// Wave w owns M-tile w (rows 16w..16w+15 of `diffusion`). K-tiles 0..7 live
// in registers (64 VGPRs/lane, bf16); K-tiles 8..15 live in LDS pre-swizzled
// into WMMA A-fragment order (256KB of the 320KB CDNA5 WGP LDS). Each Euler
// step's matvec is 16 chained v_wmma_f32_16x16x32_bf16 ops.
// ---------------------------------------------------------------------------
__global__ __launch_bounds__(1024, 1)
void attractor_scan_kernel(const float* __restrict__ embed,
                           const float* __restrict__ diffusion,
                           const float* __restrict__ beta_p,
                           const float* __restrict__ gamma_p,
                           const float* __restrict__ traw_p,
                           const int*   __restrict__ token_ids,
                           float* __restrict__ ws_state,   // [kD]
                           float* __restrict__ ws_ctx,     // [kD]
                           float* __restrict__ ws_temp)    // [1]
{
  extern __shared__ char smem_raw[];
  __bf16* sA    = (__bf16*)smem_raw;                    // streamed A half
  float*  sS    = (float*)(smem_raw + kABytes);         // fast state
  float*  sSlow = sS + kD;
  float*  sSig  = sSlow + kD;
  float*  sY    = sSig + kD;                            // matvec result
  float*  sE    = sY + kD;                              // staged embed row
  __bf16* sSbf  = (__bf16*)(sE + kD);                   // bf16 shadow of sS
  float*  sRed  = (float*)(sSbf + kD);                  // broadcast scalars

  const int tid    = threadIdx.x;
  const int wave   = tid >> 5;
  const int lane   = tid & 31;
  const int mrow   = lane & 15;
  const int khalfA = lane >> 4;              // A-frag K half (offset 16)
  const int kbB    = (lane >= 16) ? 16 : 0;  // B-frag K half (offset 16)

  const float beta  = *beta_p;
  const float gamma = *gamma_p;

  if (tid < kD) { sS[tid] = 0.f; sSlow[tid] = 0.f; sSbf[tid] = (__bf16)0.f; }

  // ---- one-time gather of A fragments (bf16, contiguous k = 16*half + s, matching B) ----
  // K-tiles 0..7 -> registers; K-tiles 8..15 -> LDS, stored exactly as the
  // v16bf this (wave,lane) will later stream back (layout trivially right).
  v16bf afrag[8];
  const float* Arow = diffusion + (long)(wave * 16 + mrow) * kD;
#pragma unroll
  for (int kt = 0; kt < 8; ++kt) {
#pragma unroll
    for (int v = 0; v < 8; ++v) {
      const int kb = kt * 32 + v * 2 + khalfA * 16;
      afrag[kt][2 * v]     = (__bf16)Arow[kb];
      afrag[kt][2 * v + 1] = (__bf16)Arow[kb + 1];
    }
  }
#pragma unroll
  for (int ktl = 0; ktl < 8; ++ktl) {
    v16bf tmp;
#pragma unroll
    for (int v = 0; v < 8; ++v) {
      const int kb = (ktl + 8) * 32 + v * 2 + khalfA * 16;
      tmp[2 * v]     = (__bf16)Arow[kb];
      tmp[2 * v + 1] = (__bf16)Arow[kb + 1];
    }
    *(v16bf*)(sA + ((wave * 8 + ktl) * 32 + lane) * 16) = tmp;
  }
  __syncthreads();

  for (int t = 0; t < kT; ++t) {
    const int tok = token_ids[t];
    const float* erow = embed + (long)tok * kD;
    if (tid < kD) sE[tid] = erow[tid];
    if (t + 1 < kT && tid < 16) {
      const int ntok = token_ids[t + 1];
      __builtin_prefetch(embed + (long)ntok * kD + tid * 32, 0, 3);
    }
    __syncthreads();

    // ---- parallel reductions: embed mean/var, |fast|^2, |combined|^2 ----
    if (wave == 0) {
      float s = 0.f, s2 = 0.f;
#pragma unroll
      for (int j = 0; j < 16; ++j) { float e = sE[lane + 32 * j]; s += e; s2 += e * e; }
      s = waveAllSum(s); s2 = waveAllSum(s2);
      if (lane == 0) {
        float m = s * (1.f / kD);
        sRed[0] = m;
        sRed[1] = s2 * (1.f / kD) - m * m;
      }
    } else if (wave == 1) {
      float f2 = 0.f;
#pragma unroll
      for (int j = 0; j < 16; ++j) { float x = sS[lane + 32 * j]; f2 += x * x; }
      f2 = waveAllSum(f2);
      if (lane == 0) sRed[2] = f2;
    } else if (wave == 2) {
      float c2 = 0.f;
#pragma unroll
      for (int j = 0; j < 16; ++j) {
        int i = lane + 32 * j;
        float x = sS[i] + kWSlow * sSlow[i];
        c2 += x * x;
      }
      c2 = waveAllSum(c2);
      if (lane == 0) sRed[3] = c2;
    }
    __syncthreads();

    // ---- signal = normalize(base_signal(embed_row) + gamma*ctx) ----
    if (tid < kD) {
      const float m = sRed[0], var = sRed[1];
      const float fn = sqrtf(sRed[2]);
      const float cn = sqrtf(sRed[3]);
      const float b  = (sE[tid] - m) / sqrtf(var + kLnEps);
      const float bn = fmaxf(sqrtf((float)kD * var / (var + kLnEps)), 1e-12f);
      const float base = b / bn;
      const float comb = sS[tid] + kWSlow * sSlow[tid];
      const float ctx  = (fn > 1e-8f) ? (sS[tid] / (fn + kEps))
                        : ((cn > 1e-8f) ? (comb / (cn + kEps)) : 0.f);
      sSig[tid] = base + gamma * ctx;
    }
    __syncthreads();
    if (wave == 0) {
      float n2 = 0.f;
#pragma unroll
      for (int j = 0; j < 16; ++j) { float x = sSig[lane + 32 * j]; n2 += x * x; }
      n2 = waveAllSum(n2);
      if (lane == 0) sRed[4] = sqrtf(n2);
    }
    __syncthreads();
    if (tid < kD) sSig[tid] = sSig[tid] / (sRed[4] + kEps);
    __syncthreads();

    // ---- 4 Euler steps ----
    for (int st = 0; st < kSteps; ++st) {
      if (wave == 0) {
        float s = 0.f;
#pragma unroll
        for (int j = 0; j < 16; ++j) s += sS[lane + 32 * j];
        s = waveAllSum(s);
        if (lane == 0) sRed[5] = s * (1.f / kD);
      }
      // matvec y = diffusion @ s via WMMA (all 32 waves, full EXEC).
      // B-fragment = 16 contiguous bf16 of the state shadow (broadcast cols).
      v8f acc = {};
#pragma unroll
      for (int kt = 0; kt < 8; ++kt) {
        v16bf bfrag = *(const v16bf*)(sSbf + kt * 32 + kbB);
        acc = __builtin_amdgcn_wmma_f32_16x16x32_bf16(
            false, afrag[kt], false, bfrag, (short)0, acc, false, false);
      }
#pragma unroll
      for (int ktl = 0; ktl < 8; ++ktl) {
        v16bf astream = *(const v16bf*)(sA + ((wave * 8 + ktl) * 32 + lane) * 16);
        v16bf bfrag   = *(const v16bf*)(sSbf + (ktl + 8) * 32 + kbB);
        acc = __builtin_amdgcn_wmma_f32_16x16x32_bf16(
            false, astream, false, bfrag, (short)0, acc, false, false);
      }
      // D layout: lane 0 holds M=0..7 (N=0 column), lane 16 holds M=8..15
      if (lane == 0) {
#pragma unroll
        for (int j = 0; j < 8; ++j) sY[wave * 16 + j] = acc[j];
      } else if (lane == 16) {
#pragma unroll
        for (int j = 0; j < 8; ++j) sY[wave * 16 + 8 + j] = acc[j];
      }
      __syncthreads();
      if (tid < kD) {
        const float s = sS[tid];
        const float c = s - sRed[5];
        const float drift = sY[tid] + kCubic * c * c * c + beta * sSig[tid];
        float ns = s + kDt * drift;
        ns = isfinite(ns) ? ns : 0.f;          // nan_to_num(nan=0, +/-inf=0)
        ns = fminf(fmaxf(ns, -80.f), 80.f);
        sS[tid]   = ns;
        sSbf[tid] = (__bf16)ns;                // keep bf16 shadow in sync
      }
      __syncthreads();
    }

    if (tid < kD) sSlow[tid] = kAlpha * sSlow[tid] + (1.f - kAlpha) * sS[tid];
    __syncthreads();
  }

  // ---- finalize: state = fast + 0.5*slow, ctx, temperature ----
  if (wave == 0) {
    float f2 = 0.f;
#pragma unroll
    for (int j = 0; j < 16; ++j) { float x = sS[lane + 32 * j]; f2 += x * x; }
    f2 = waveAllSum(f2);
    if (lane == 0) sRed[2] = f2;
  } else if (wave == 1) {
    float c2 = 0.f;
#pragma unroll
    for (int j = 0; j < 16; ++j) {
      int i = lane + 32 * j;
      float x = sS[i] + kWSlow * sSlow[i];
      c2 += x * x;
    }
    c2 = waveAllSum(c2);
    if (lane == 0) sRed[3] = c2;
  }
  __syncthreads();
  if (tid < kD) {
    const float fn = sqrtf(sRed[2]);
    const float cn = sqrtf(sRed[3]);
    const float comb = sS[tid] + kWSlow * sSlow[tid];
    const float ctx = (fn > 1e-8f) ? (sS[tid] / (fn + kEps))
                     : ((cn > 1e-8f) ? (comb / (cn + kEps)) : 0.f);
    ws_ctx[tid]   = ctx;
    ws_state[tid] = comb;
  }
  if (tid == 0) {
    const float x = *traw_p;
    const float sp = (x > 20.f) ? x : log1pf(expf(x));
    ws_temp[0] = fmaxf(sp, 1e-6f);
  }
}

// ---------------------------------------------------------------------------
// Vocab logits: one wave32 per vocab row; HBM-bandwidth bound (64 MB embed).
// ---------------------------------------------------------------------------
__global__ __launch_bounds__(256)
void attractor_logits_kernel(const float* __restrict__ embed,
                             const float* __restrict__ gamma_p,
                             const float* __restrict__ ws_state,
                             const float* __restrict__ ws_ctx,
                             const float* __restrict__ ws_temp,
                             float* __restrict__ out, int V)
{
  __shared__ float sState[kD];
  __shared__ float sCtx[kD];
  const int tid  = threadIdx.x;
  const int wave = tid >> 5;
  const int lane = tid & 31;
  for (int i = tid; i < kD; i += 256) { sState[i] = ws_state[i]; sCtx[i] = ws_ctx[i]; }
  __syncthreads();

  const float gamma = *gamma_p;
  const float temp  = ws_temp[0];
  const int row = blockIdx.x * 8 + wave;
  if (row >= V) return;

  const float* erow = embed + (long)row * kD;
  float e[16];
  float s = 0.f, s2 = 0.f;
#pragma unroll
  for (int j = 0; j < 16; ++j) {
    e[j] = erow[lane + 32 * j];
    s += e[j]; s2 += e[j] * e[j];
  }
  s = waveAllSum(s); s2 = waveAllSum(s2);
  const float m   = s * (1.f / kD);
  const float var = s2 * (1.f / kD) - m * m;
  const float inv = 1.f / sqrtf(var + kLnEps);
  const float bn  = fmaxf(sqrtf((float)kD * var / (var + kLnEps)), 1e-12f);

  float sr[16];
  float n2 = 0.f;
#pragma unroll
  for (int j = 0; j < 16; ++j) {
    const int i = lane + 32 * j;
    const float b = (e[j] - m) * inv / bn;
    sr[j] = b + gamma * sCtx[i];
    n2 += sr[j] * sr[j];
  }
  n2 = waveAllSum(n2);
  const float invn = 1.f / (sqrtf(n2) + kEps);

  float d2 = 0.f;
#pragma unroll
  for (int j = 0; j < 16; ++j) {
    const int i = lane + 32 * j;
    const float diff = sState[i] - sr[j] * invn;
    d2 += diff * diff;
  }
  d2 = waveAllSum(d2);

  if (lane == 0) {
    float logit = -sqrtf(d2) / temp;
    if (isnan(logit)) logit = 0.f;
    else if (isinf(logit)) logit = (logit > 0.f) ? 0.f : -10000.f;
    out[row] = logit;
  }
}

extern "C" void kernel_launch(void* const* d_in, const int* in_sizes, int n_in,
                              void* d_out, int out_size, void* d_ws, size_t ws_size,
                              hipStream_t stream) {
  (void)in_sizes; (void)n_in; (void)ws_size;
  const float* embed     = (const float*)d_in[0];
  const float* diffusion = (const float*)d_in[1];
  const float* beta_p    = (const float*)d_in[2];
  const float* gamma_p   = (const float*)d_in[3];
  const float* traw_p    = (const float*)d_in[4];
  const int*   token_ids = (const int*)d_in[5];

  float* ws       = (float*)d_ws;
  float* ws_state = ws;
  float* ws_ctx   = ws + kD;
  float* ws_temp  = ws + 2 * kD;

  // Opt in to the large CDNA5 per-WGP LDS (267KB dynamic shared).
  (void)hipFuncSetAttribute((const void*)attractor_scan_kernel,
                            hipFuncAttributeMaxDynamicSharedMemorySize,
                            kSmemSize);

  attractor_scan_kernel<<<1, 1024, kSmemSize, stream>>>(
      embed, diffusion, beta_p, gamma_p, traw_p, token_ids,
      ws_state, ws_ctx, ws_temp);

  const int V = out_size;                 // 32000
  const int blocks = (V + 7) / 8;         // one wave32 per vocab row
  attractor_logits_kernel<<<blocks, 256, 0, stream>>>(
      embed, gamma_p, ws_state, ws_ctx, ws_temp, (float*)d_out, V);
}
